// TransformerBlock_50165218017587
// MI455X (gfx1250) — compile-verified, matrix operand order checked by reading, not run on hardware
//
#include <hip/hip_runtime.h>
#include <math.h>

typedef __attribute__((ext_vector_type(16))) __bf16 v16bf;
typedef __attribute__((ext_vector_type(8)))  float  v8f;
typedef unsigned short u16;
typedef unsigned int   u32;
typedef __attribute__((ext_vector_type(4)))  u32    v4u;   // native vector (not HIP class)

#define GLOBAL_AS __attribute__((address_space(1)))

// ---------- bf16 helpers (storage type: u16) ----------
__device__ inline u16 f32_to_bf16(float f) {
  u32 u = __builtin_bit_cast(u32, f);
  u32 r = u + 0x7FFFu + ((u >> 16) & 1u);   // round-to-nearest-even
  return (u16)(r >> 16);
}
__device__ inline float bf16_to_f32(u16 h) {
  u32 u = ((u32)h) << 16;
  return __builtin_bit_cast(float, u);
}

// ---------- explicit global-address-space accessors (force VGLOBAL, not FLAT) ----------
__device__ inline float gload_f32(const float* p) { return *(const GLOBAL_AS float*)p; }
__device__ inline u16   gload_u16(const u16* p)   { return *(const GLOBAL_AS u16*)p; }
__device__ inline void  gstore_f32(float* p, float v) { *(GLOBAL_AS float*)p = v; }
__device__ inline void  gstore_u16(u16* p, u16 v)     { *(GLOBAL_AS u16*)p = v; }

// Build a 16x(bf16) WMMA fragment from two contiguous 8-element (16B) runs.
// Per ISA 7.12.2, lane needs K = kbase+{0..7} (elems 0-7) and kbase+16+{0..7}
// (elems 8-15), where kbase = (lane>>4)*8. Caller passes ptr = row + kbase.
struct bf16x16_bits { v4u lo, hi; };
__device__ inline v16bf load_frag_g(const u16* p) {   // global: 2x global_load_b128
  const GLOBAL_AS v4u* g = (const GLOBAL_AS v4u*)(const GLOBAL_AS u16*)p;
  bf16x16_bits t;
  t.lo = g[0];
  t.hi = g[2];   // +32 bytes
  return __builtin_bit_cast(v16bf, t);
}
__device__ inline v16bf load_frag_lds(const u16* p) { // LDS: 2x ds_load_b128
  bf16x16_bits t;
  t.lo = *(const v4u*)p;
  t.hi = *(const v4u*)(p + 16);
  return __builtin_bit_cast(v16bf, t);
}

__device__ inline v8f wmma_bf16(v16bf a, v16bf b, v8f c) {
  return __builtin_amdgcn_wmma_f32_16x16x32_bf16(false, a, false, b, (short)0, c, false, false);
}

// ---------- f32 -> bf16 conversion (plain) ----------
__global__ void cvt_f32_bf16_kernel(const float* __restrict__ src, u16* __restrict__ dst, int n) {
  int i = blockIdx.x * blockDim.x + threadIdx.x;
  if (i < n) gstore_u16(dst + i, f32_to_bf16(gload_f32(src + i)));
}

// ---------- f32 -> bf16 conversion with transpose ----------
// src: [H, K, hd] f32 (plain [K,N] is H=1, hd=N). dst: [H*hd, K] bf16 row-major.
__global__ void cvt_transpose_kernel(const float* __restrict__ src, u16* __restrict__ dst,
                                     int K, int hd, int total) {
  int i = blockIdx.x * blockDim.x + threadIdx.x;
  if (i >= total) return;
  int r  = i / K;          // output row = h*hd + e
  int kk = i - r * K;
  int h  = r / hd, e = r - h * hd;
  gstore_u16(dst + i, f32_to_bf16(gload_f32(src + ((size_t)h * K + kk) * (size_t)hd + e)));
}

// ---------- LayerNorm (one row of D per block), writes bf16 ----------
__global__ __launch_bounds__(256)
void layernorm_bf16_kernel(const float* __restrict__ x, const float* __restrict__ g,
                           const float* __restrict__ b, u16* __restrict__ out, int D) {
  const int row = blockIdx.x;
  const float* xr = x + (size_t)row * D;
  float s = 0.f, s2 = 0.f;
  for (int i = threadIdx.x; i < D; i += 256) { float v = gload_f32(xr + i); s += v; s2 += v * v; }
  for (int m = 16; m; m >>= 1) { s += __shfl_xor(s, m, 32); s2 += __shfl_xor(s2, m, 32); }
  __shared__ float ss[8], ss2[8];
  int w = threadIdx.x >> 5;
  if ((threadIdx.x & 31) == 0) { ss[w] = s; ss2[w] = s2; }
  __syncthreads();
  float ts = 0.f, ts2 = 0.f;
  for (int i = 0; i < 8; ++i) { ts += ss[i]; ts2 += ss2[i]; }
  float mu = ts / (float)D;
  float var = ts2 / (float)D - mu * mu;
  float rstd = rsqrtf(var + 1e-5f);
  for (int i = threadIdx.x; i < D; i += 256) {
    float v = (gload_f32(xr + i) - mu) * rstd * gload_f32(g + i) + gload_f32(b + i);
    gstore_u16(out + (size_t)row * D + i, f32_to_bf16(v));
  }
}

// ---------- Generic bf16 WMMA GEMM: C[M,N] = act(A[M,K] * Bt[N,K]^T + bias) ----------
// No LDS: all fragments are direct global_load_b128 (weights pre-transposed,
// activations row-major bf16; everything L2-resident on MI455X's 192MB L2).
// Block = 128 threads = 4 waves; wave tile = 32 rows x 64 cols (2 A-frags share
// each B-frag). All 12 loads of a k-step are issued before the 8 WMMAs so the
// backend can clause them and stagger s_wait_loadcnt (memory-level parallelism).
// ACT: 0=none, 1=tanh-GELU. scale folds 1/sqrt(D) into Q.
template<int ACT, bool OUT_BF16, bool OUT_F32, bool RESID>
__global__ __launch_bounds__(128)
void gemm_bf16_kernel(const u16* __restrict__ A, int lda,
                      const u16* __restrict__ Bt,
                      const float* __restrict__ bias,
                      u16* __restrict__ outB, float* __restrict__ outF,
                      const float* __restrict__ resid,
                      int N, int K, float scale) {
  const int lane = threadIdx.x & 31, wave = threadIdx.x >> 5;
  const int half = lane >> 4, nl = lane & 15;
  const int kb = half * 8;
  const int m0 = blockIdx.y * 128 + wave * 32;
  const int n0 = blockIdx.x * 64;

  v8f acc[2][4];
  for (int r = 0; r < 2; ++r)
    for (int t = 0; t < 4; ++t)
      for (int j = 0; j < 8; ++j) acc[r][t][j] = 0.f;

  const u16* a0 = A + (size_t)(m0 + nl) * lda + kb;
  const u16* a1 = A + (size_t)(m0 + 16 + nl) * lda + kb;
  const u16* bp0 = Bt + (size_t)(n0 +  0 + nl) * K + kb;
  const u16* bp1 = Bt + (size_t)(n0 + 16 + nl) * K + kb;
  const u16* bp2 = Bt + (size_t)(n0 + 32 + nl) * K + kb;
  const u16* bp3 = Bt + (size_t)(n0 + 48 + nl) * K + kb;

#pragma unroll 2
  for (int k0 = 0; k0 < K; k0 += 32) {
    // issue all loads first -> one clause, staggered waits
    v16bf af0 = load_frag_g(a0 + k0);
    v16bf af1 = load_frag_g(a1 + k0);
    v16bf b0  = load_frag_g(bp0 + k0);
    v16bf b1  = load_frag_g(bp1 + k0);
    v16bf b2  = load_frag_g(bp2 + k0);
    v16bf b3  = load_frag_g(bp3 + k0);
    acc[0][0] = wmma_bf16(af0, b0, acc[0][0]);
    acc[1][0] = wmma_bf16(af1, b0, acc[1][0]);
    acc[0][1] = wmma_bf16(af0, b1, acc[0][1]);
    acc[1][1] = wmma_bf16(af1, b1, acc[1][1]);
    acc[0][2] = wmma_bf16(af0, b2, acc[0][2]);
    acc[1][2] = wmma_bf16(af1, b2, acc[1][2]);
    acc[0][3] = wmma_bf16(af0, b3, acc[0][3]);
    acc[1][3] = wmma_bf16(af1, b3, acc[1][3]);
  }

  for (int r = 0; r < 2; ++r)
    for (int t = 0; t < 4; ++t) {
      int n = n0 + t * 16 + nl;
      float bv = bias ? gload_f32(bias + n) : 0.f;
      for (int j = 0; j < 8; ++j) {
        int m = m0 + r * 16 + j + half * 8;
        float v = acc[r][t][j] + bv;
        if (ACT == 1) {
          const float c0 = 0.7978845608028654f; // sqrt(2/pi)
          float u = c0 * (v + 0.044715f * v * v * v);
          v = 0.5f * v * (1.f + tanhf(u));
        }
        v *= scale;
        size_t idx = (size_t)m * N + n;
        if (OUT_BF16) gstore_u16(outB + idx, f32_to_bf16(v));
        if (OUT_F32)  gstore_f32(outF + idx, v + (RESID ? gload_f32(resid + idx) : 0.f));
      }
    }
}

// ---------- v = vd @ Wvu + bvu (K = 32, tiny) -> writes V TRANSPOSED ----------
// vt layout: [B, H, HD, S] so attention PV B-fragments are key-contiguous.
__global__ __launch_bounds__(256)
void vu_kernel(const u16* __restrict__ vd, const u16* __restrict__ Wvu,
               const float* __restrict__ bvu, u16* __restrict__ vt) {
  const int S = 2048, H = 16;
  const int h = blockIdx.y;
  const int m = blockIdx.x * 4 + (threadIdx.x >> 6);
  const int e = threadIdx.x & 63;
  float acc = gload_f32(bvu + h * 64 + e);
  const u16* vr = vd + (size_t)m * 512 + h * 32;
  const u16* wr = Wvu + (size_t)h * (32 * 64) + e;
  for (int r = 0; r < 32; ++r)
    acc += bf16_to_f32(gload_u16(vr + r)) * bf16_to_f32(gload_u16(wr + r * 64));
  int b = m >> 11, s = m & (S - 1);
  gstore_u16(vt + ((size_t)((b * H + h) * 64 + e)) * S + s, f32_to_bf16(acc));
}

// ---------- Flash attention (reversed-causal: keys t >= query s allowed) ----------
// q/k bf16 [B,S,H*HD]; vt bf16 [B,H,HD,S]; writes x2 = attn_out + x (f32).
// All K/V/Q fragments: direct global_load_b128, batched ahead of the WMMAs.
// Only P re-swizzles via a per-wave LDS tile (scalar stores, ds_load_b128).
// No block barriers. The diagonal kv tile is peeled: steady-state has no mask.
__global__ __launch_bounds__(128)
void attention_kernel(const u16* __restrict__ q, const u16* __restrict__ k,
                      const u16* __restrict__ vt, const float* __restrict__ x,
                      float* __restrict__ x2) {
  constexpr int S = 2048, D = 1024, H = 16;
  __shared__ u16 Ps[4][16][80];   // per-wave P tile, 160B row stride (16B aligned)
  const int lane = threadIdx.x & 31, wave = threadIdx.x >> 5;
  const int half = lane >> 4, nl = lane & 15;
  const int kb = half * 8;
  const int qbase = blockIdx.x * 64;
  const int h = blockIdx.y, b = blockIdx.z;

  // Q fragments (HD=64 -> two k-steps); lane row m = nl
  const u16* qp = q + ((size_t)(b * S + qbase + wave * 16 + nl)) * D + h * 64 + kb;
  v16bf qf[2] = { load_frag_g(qp), load_frag_g(qp + 32) };

  // per-tile K row pointers (advance by 64 rows per kv tile)
  const u16* kp[4];
  for (int t = 0; t < 4; ++t)
    kp[t] = k + ((size_t)(b * S + qbase + t * 16 + nl)) * D + h * 64 + kb;
  // Vt row pointers (row = hd, advance along S per kv tile)
  const u16* vp[4];
  for (int t = 0; t < 4; ++t)
    vp[t] = vt + ((size_t)((b * H + h) * 64 + t * 16 + nl)) * S + kb;

  float mrow[8], lrow[8];
  v8f oacc[4];
  for (int j = 0; j < 8; ++j) { mrow[j] = -__builtin_inff(); lrow[j] = 0.f; }
  for (int t = 0; t < 4; ++t)
    for (int j = 0; j < 8; ++j) oacc[t][j] = 0.f;

  constexpr size_t kstep = (size_t)64 * D;   // K pointer advance per kv tile

  auto kv_tile = [&](int kt, bool do_mask) {
    // ---- S = Q * K^T (Q pre-scaled by 1/sqrt(D)); loads batched per f-step ----
    v8f sacc[4];
    for (int t = 0; t < 4; ++t)
      for (int j = 0; j < 8; ++j) sacc[t][j] = 0.f;
    for (int f = 0; f < 2; ++f) {
      v16bf k0f = load_frag_g(kp[0] + f * 32);
      v16bf k1f = load_frag_g(kp[1] + f * 32);
      v16bf k2f = load_frag_g(kp[2] + f * 32);
      v16bf k3f = load_frag_g(kp[3] + f * 32);
      sacc[0] = wmma_bf16(qf[f], k0f, sacc[0]);
      sacc[1] = wmma_bf16(qf[f], k1f, sacc[1]);
      sacc[2] = wmma_bf16(qf[f], k2f, sacc[2]);
      sacc[3] = wmma_bf16(qf[f], k3f, sacc[3]);
    }
    for (int t = 0; t < 4; ++t) kp[t] += kstep;

    // ---- (peeled diagonal tile only) mask: key < query -> -inf ----
    if (do_mask) {
      for (int t = 0; t < 4; ++t) {
        int key = kt + t * 16 + nl;
        for (int j = 0; j < 8; ++j) {
          int qr = qbase + wave * 16 + j + half * 8;
          if (key < qr) sacc[t][j] = -__builtin_inff();
        }
      }
    }

    // ---- row max ----
    float newm[8];
    for (int j = 0; j < 8; ++j) {
      float nm = fmaxf(fmaxf(sacc[0][j], sacc[1][j]), fmaxf(sacc[2][j], sacc[3][j]));
      newm[j] = fmaxf(mrow[j], nm);
    }
    for (int j = 0; j < 8; ++j)
      for (int m2 = 8; m2; m2 >>= 1)
        newm[j] = fmaxf(newm[j], __shfl_xor(newm[j], m2, 32));

    // ---- P = exp(S - m), online rescale ----
    float fac[8], rsum[8];
    for (int j = 0; j < 8; ++j) {
      float sm = (newm[j] == -__builtin_inff()) ? 0.f : newm[j];
      fac[j] = __expf(mrow[j] - sm);
      float rs = 0.f;
      for (int t = 0; t < 4; ++t) {
        float p = __expf(sacc[t][j] - sm);
        rs += p;
        Ps[wave][j + half * 8][t * 16 + nl] = f32_to_bf16(p);
      }
      for (int m2 = 8; m2; m2 >>= 1) rs += __shfl_xor(rs, m2, 32);
      rsum[j] = rs;
    }
    for (int j = 0; j < 8; ++j) {
      for (int t = 0; t < 4; ++t) oacc[t][j] *= fac[j];
      lrow[j] = lrow[j] * fac[j] + rsum[j];
      mrow[j] = newm[j];
    }

    // ---- O += P * V (P via per-wave LDS, V via transposed global) ----
    for (int f = 0; f < 2; ++f) {
      v16bf pf = load_frag_lds(&Ps[wave][nl][f * 32 + kb]);
      v16bf v0 = load_frag_g(vp[0] + kt + f * 32);
      v16bf v1 = load_frag_g(vp[1] + kt + f * 32);
      v16bf v2 = load_frag_g(vp[2] + kt + f * 32);
      v16bf v3 = load_frag_g(vp[3] + kt + f * 32);
      oacc[0] = wmma_bf16(pf, v0, oacc[0]);
      oacc[1] = wmma_bf16(pf, v1, oacc[1]);
      oacc[2] = wmma_bf16(pf, v2, oacc[2]);
      oacc[3] = wmma_bf16(pf, v3, oacc[3]);
    }
  };

  kv_tile(qbase, true);                                  // diagonal tile (masked)
  for (int kt = qbase + 64; kt < S; kt += 64)
    kv_tile(kt, false);                                  // steady state (no mask)

  // epilogue: normalize, add residual x, write f32
  for (int t = 0; t < 4; ++t)
    for (int j = 0; j < 8; ++j) {
      int row = qbase + wave * 16 + j + half * 8;
      int col = h * 64 + t * 16 + nl;
      float ov = (lrow[j] > 0.f) ? oacc[t][j] / lrow[j] : 0.f;
      size_t idx = ((size_t)(b * S + row)) * D + col;
      gstore_f32(x2 + idx, ov + gload_f32(x + idx));
    }
}

// ---------------- host launcher ----------------
extern "C" void kernel_launch(void* const* d_in, const int* in_sizes, int n_in,
                              void* d_out, int out_size, void* d_ws, size_t ws_size,
                              hipStream_t stream) {
  (void)in_sizes; (void)n_in; (void)out_size; (void)ws_size;
  const float* x    = (const float*)d_in[0];
  const float* ln1g = (const float*)d_in[1];
  const float* ln1b = (const float*)d_in[2];
  const float* Wq   = (const float*)d_in[3];
  const float* bq   = (const float*)d_in[4];
  const float* Wk   = (const float*)d_in[5];
  const float* bk   = (const float*)d_in[6];
  const float* Wvd  = (const float*)d_in[7];
  const float* bvd  = (const float*)d_in[8];
  const float* Wvu  = (const float*)d_in[9];
  const float* bvu  = (const float*)d_in[10];
  const float* ln2g = (const float*)d_in[11];
  const float* ln2b = (const float*)d_in[12];
  const float* W1   = (const float*)d_in[13];
  const float* b1   = (const float*)d_in[14];
  const float* W2   = (const float*)d_in[15];
  const float* b2   = (const float*)d_in[16];

  const int Bc = 2, S = 2048, D = 1024, H = 16, HD = 64, R = 32, F = 4096;
  const int M = Bc * S;

  char* ws = (char*)d_ws;
  size_t off = 0;
  auto alloc = [&](size_t bytes) -> char* {
    char* p = ws + off;
    off += (bytes + 255) & ~(size_t)255;
    return p;
  };
  u16*   hbf  = (u16*)alloc((size_t)M * D * 2);
  u16*   h2bf = (u16*)alloc((size_t)M * D * 2);
  u16*   qbf  = (u16*)alloc((size_t)M * D * 2);
  u16*   kbf  = (u16*)alloc((size_t)M * D * 2);
  u16*   vtbf = (u16*)alloc((size_t)M * D * 2);        // [B,H,HD,S]
  u16*   vdbf = (u16*)alloc((size_t)M * (H * R) * 2);
  u16*   f1bf = (u16*)alloc((size_t)M * F * 2);
  float* x2   = (float*)alloc((size_t)M * D * 4);
  u16*   Wqt  = (u16*)alloc((size_t)H * D * HD * 2);   // [H*HD, D]
  u16*   Wkt  = (u16*)alloc((size_t)H * D * HD * 2);   // [H*HD, D]
  u16*   Wvdt = (u16*)alloc((size_t)H * D * R * 2);    // [H*R, D]
  u16*   Wvub = (u16*)alloc((size_t)H * R * HD * 2);   // plain [H,R,HD]
  u16*   W1t  = (u16*)alloc((size_t)D * F * 2);        // [F, D]
  u16*   W2t  = (u16*)alloc((size_t)F * D * 2);        // [D, F]

  auto cvtT = [&](const float* s, u16* d, int K, int hd, int total) {
    cvt_transpose_kernel<<<(total + 255) / 256, 256, 0, stream>>>(s, d, K, hd, total);
  };
  cvtT(Wq,  Wqt,  D, HD, H * D * HD);
  cvtT(Wk,  Wkt,  D, HD, H * D * HD);
  cvtT(Wvd, Wvdt, D, R,  H * D * R);
  cvt_f32_bf16_kernel<<<(H * R * HD + 255) / 256, 256, 0, stream>>>(Wvu, Wvub, H * R * HD);
  cvtT(W1,  W1t,  D, F,  D * F);   // [D,F] -> [F,D]
  cvtT(W2,  W2t,  F, D,  F * D);   // [F,D] -> [D,F]

  layernorm_bf16_kernel<<<M, 256, 0, stream>>>(x, ln1g, ln1b, hbf, D);

  dim3 gqk(D / 64, M / 128);
  // Q: fold 1/sqrt(D)=1/32 into the stored q
  gemm_bf16_kernel<0, true, false, false><<<gqk, 128, 0, stream>>>(
      hbf, D, Wqt, bq, qbf, nullptr, nullptr, D, D, 1.f / 32.f);
  gemm_bf16_kernel<0, true, false, false><<<gqk, 128, 0, stream>>>(
      hbf, D, Wkt, bk, kbf, nullptr, nullptr, D, D, 1.f);
  dim3 gvd((H * R) / 64, M / 128);
  gemm_bf16_kernel<0, true, false, false><<<gvd, 128, 0, stream>>>(
      hbf, D, Wvdt, bvd, vdbf, nullptr, nullptr, H * R, D, 1.f);

  vu_kernel<<<dim3(M / 4, H), 256, 0, stream>>>(vdbf, Wvub, bvu, vtbf);

  attention_kernel<<<dim3(S / 64, H, Bc), 128, 0, stream>>>(qbf, kbf, vtbf, x, x2);

  layernorm_bf16_kernel<<<M, 256, 0, stream>>>(x2, ln2g, ln2b, h2bf, D);

  dim3 gf1(F / 64, M / 128);
  gemm_bf16_kernel<1, true, false, false><<<gf1, 128, 0, stream>>>(
      h2bf, D, W1t, b1, f1bf, nullptr, nullptr, F, D, 1.f);
  dim3 gf2(D / 64, M / 128);
  gemm_bf16_kernel<0, false, true, true><<<gf2, 128, 0, stream>>>(
      f1bf, F, W2t, b2, nullptr, (float*)d_out, x2, D, F, 1.f);
}
